// Attention_8504035246380
// MI455X (gfx1250) — matrix operand order repaired and checked by reading, not compiled, not run on hardware
//
#include <hip/hip_runtime.h>
#include <hip/hip_bf16.h>

// ---------------------------------------------------------------------------
// Causal attention block (PBrelax == plain softmax(QK^T/sqrt(HD)) after the
// stop_gradient identity) for B=2, T=2048, C=1024, H=16, HD=64 on gfx1250.
// Matmuls: v_wmma_f32_16x16x32_bf16.  K/V tile staging: TDM tensor_load_to_lds.
// ---------------------------------------------------------------------------

typedef __attribute__((ext_vector_type(16))) __bf16    v16bf;
typedef __attribute__((ext_vector_type(8)))  float     v8f;
typedef __attribute__((ext_vector_type(4)))  unsigned  u32x4;
typedef __attribute__((ext_vector_type(8)))  unsigned  u32x8;

#define B_   2
#define T_   2048
#define C_   1024
#define H_   16
#define HD_  64
#define M_   (B_ * T_)          // 4096 rows for all projections
#define SM_SCALE 0.125f          // 1/sqrt(64)

static __device__ __forceinline__ __bf16 f2bf(float f) {
    union { float f; unsigned u; } v; v.f = f;
    unsigned r = v.u + 0x7FFFu + ((v.u >> 16) & 1u);   // round-to-nearest-even
    unsigned short h = (unsigned short)(r >> 16);
    return __builtin_bit_cast(__bf16, h);
}

static __device__ __forceinline__ v8f wmma_bf16(v16bf a, v16bf b, v8f c) {
    return __builtin_amdgcn_wmma_f32_16x16x32_bf16(
        /*neg_a=*/false, a, /*neg_b=*/false, b,
        /*c_mod=*/(short)0, c, /*reuse_a=*/false, /*reuse_b=*/false);
}

// ---------------------------------------------------------------------------
// TDM: async-DMA a contiguous run of `nelem` bf16 elements from global memory
// into LDS at byte offset `lds_off`.  Builds the D# descriptor (ISA ch.8):
//   group0: count=1 | lds_addr | global_addr[56:0] | type=2 ("image")
//   group1: data_size=1 (2B), 1-D tile: tile_dim0=nelem, tensor_dim0=nelem,
//           tensor_dim0_stride=nelem, everything else 0 (no cluster multicast).
// Completion is tracked with TENSORcnt (s_wait_tensorcnt).
// ---------------------------------------------------------------------------
static __device__ __forceinline__ void tdm_load_to_lds(unsigned lds_off,
                                                       const void* gptr,
                                                       unsigned nelem) {
    unsigned long long ga = (unsigned long long)gptr;
    u32x4 g0;
    g0[0] = 0x1u;                                      // count=1, user mode
    g0[1] = lds_off;                                   // lds_addr
    g0[2] = (unsigned)(ga & 0xFFFFFFFFu);              // global_addr[31:0]
    g0[3] = (unsigned)((ga >> 32) & 0x1FFFFFFu)        // global_addr[56:32]
            | (2u << 30);                              // type=2
    u32x8 g1;
    g1[0] = (1u << 16);                                // data_size=1 -> 2 bytes
    g1[1] = (nelem & 0xFFFFu) << 16;                   // tensor_dim0[15:0]
    g1[2] = (nelem >> 16) | (1u << 16);                // tensor_dim0[31:16] | tensor_dim1=1
    g1[3] = (nelem & 0xFFFFu) << 16;                   // tile_dim0 = nelem
    g1[4] = 0;                                         // tile_dim1/2 unused
    g1[5] = nelem;                                     // tensor_dim0_stride[31:0]
    g1[6] = 0;
    g1[7] = 0;
    asm volatile("tensor_load_to_lds %0, %1" :: "s"(g0), "s"(g1) : "memory");
}

// ---------------------------------------------------------------------------
// Block-cooperative GEMM: Y[M,N] = X[M,K] * W[K,N] + bias.
// 256 threads stage a 128x32 X tile + 32x64 W tile per k-step; each of the 8
// waves computes a 16x64 strip = 4 WMMAs per k-step off one shared A fragment.
// XMODE: 0 = f32 input, 1 = bf16.  OMODE: 0 = bf16 head-split, 1 = f32 flat.
// ---------------------------------------------------------------------------
template <int XMODE, int OMODE>
__global__ __launch_bounds__(256)
void gemm_wmma(const void* __restrict__ Xv, const float* __restrict__ W,
               const float* __restrict__ bias, void* __restrict__ dstv,
               int M, int N, int K) {
    __shared__ __bf16 Xs[128 * 32];   // 8 KB
    __shared__ __bf16 Ws[32 * 64];    // 4 KB

    const int tid  = threadIdx.x;
    const int wave = tid >> 5;
    const int lane = tid & 31;
    const int nbn  = N >> 6;
    const int m0   = (blockIdx.x / nbn) << 7;   // 128-row macro tile
    const int n0   = (blockIdx.x % nbn) << 6;   // 64-col macro tile

    const float*  Xf = (const float*)Xv;
    const __bf16* Xb = (const __bf16*)Xv;

    const int ln  = lane & 15;
    const int kbA = (lane < 16) ? 0 : 8;
    const int kbB = (lane < 16) ? 0 : 16;
    v8f acc[4] = {{}, {}, {}, {}};

    for (int kk = 0; kk < K; kk += 32) {
        if (kk + 32 < K) {
            if (XMODE == 0)
                __builtin_prefetch(&Xf[(size_t)m0 * K + kk + 32], 0, 1);
            __builtin_prefetch(&W[(size_t)(kk + 32) * N + n0], 0, 1);
        }
        // Stage X tile (128x32): id = i*256+tid -> row=id/32, col=tid&31.
#pragma unroll
        for (int i = 0; i < 16; ++i) {
            int row = (i << 3) + (tid >> 5);
            int col = tid & 31;
            if (XMODE == 0)
                Xs[row * 32 + col] = f2bf(Xf[(size_t)(m0 + row) * K + kk + col]);
            else
                Xs[row * 32 + col] = Xb[(size_t)(m0 + row) * K + kk + col];
        }
        // Stage W tile (32x64): id = i*256+tid -> row=id/64, col=id&63.
#pragma unroll
        for (int i = 0; i < 8; ++i) {
            int id = (i << 8) + tid;
            int r = id >> 6, c = id & 63;
            Ws[r * 64 + c] = f2bf(W[(size_t)(kk + r) * N + n0 + c]);
        }
        __syncthreads();

        // One A fragment per wave (rows wave*16 .. wave*16+15).
        v16bf a;
#pragma unroll
        for (int i = 0; i < 16; ++i)
            a[i] = Xs[(wave * 16 + ln) * 32 + kbB + i];
        // Four B fragments -> four WMMAs reusing A.
#pragma unroll
        for (int g = 0; g < 4; ++g) {
            v16bf b;
#pragma unroll
            for (int i = 0; i < 16; ++i)
                b[i] = Ws[(kbB + i) * 64 + g * 16 + ln];
            acc[g] = wmma_bf16(a, b, acc[g]);
        }
        __syncthreads();
    }

    // Store D fragments: lane -> col, VGPR r -> row r (+8 in upper half).
#pragma unroll
    for (int g = 0; g < 4; ++g) {
#pragma unroll
        for (int r = 0; r < 8; ++r) {
            int row = m0 + wave * 16 + r + ((lane < 16) ? 0 : 8);
            int col = n0 + g * 16 + ln;
            float v = acc[g][r] + bias[col];
            if (OMODE == 1) {
                ((float*)dstv)[(size_t)row * N + col] = v;
            } else {
                int bb = row / T_, t = row % T_;
                int h = col / HD_, hd = col % HD_;
                ((__bf16*)dstv)[(((size_t)bb * H_ + h) * T_ + t) * HD_ + hd] = f2bf(v);
            }
        }
    }
}

// ---------------------------------------------------------------------------
// Flash attention: one wave per (b*H+h, 16-query block); key blocks of 32.
// K/V tiles arrive via TDM (tensor_load_to_lds + s_wait_tensorcnt).  Online
// softmax uses xor-shuffle row reductions over the WMMA C/D VGPR layout.
// ---------------------------------------------------------------------------
__global__ __launch_bounds__(128)
void attn_fwd(const __bf16* __restrict__ Q, const __bf16* __restrict__ Kmat,
              const __bf16* __restrict__ Vmat, __bf16* __restrict__ Oatt) {
    // Per-wave LDS: K tile (32x64) + V tile (32x64) + P tile (16x32)
    __shared__ __bf16 lds[4][32 * 64 + 32 * 64 + 16 * 32];

    const int wave = threadIdx.x >> 5;
    const int lane = threadIdx.x & 31;
    const int task = blockIdx.x * 4 + wave;     // 4096 tasks total
    const int qb   = task & ((T_ / 16) - 1);    // query block 0..127
    const int bh   = task >> 7;                 // 0..31  (b*H + h)

    const __bf16* Qp = Q    + (size_t)bh * T_ * HD_;
    const __bf16* Kp = Kmat + (size_t)bh * T_ * HD_;
    const __bf16* Vp = Vmat + (size_t)bh * T_ * HD_;
    __bf16* Kt = &lds[wave][0];
    __bf16* Vt = &lds[wave][32 * 64];
    __bf16* Pt = &lds[wave][2 * 32 * 64];
    const unsigned kt_off = (unsigned)(size_t)(void*)Kt;  // LDS byte offsets
    const unsigned vt_off = (unsigned)(size_t)(void*)Vt;

    const int ln  = lane & 15;
    const int kbA = (lane < 16) ? 0 : 8;
    const int kbB = (lane < 16) ? 0 : 16;

    // Q A-fragments for the two K-halves of HD=64 (gathered from global).
    const __bf16* qrow = Qp + (size_t)(qb * 16 + ln) * HD_;
    v16bf qa0, qa1;
#pragma unroll
    for (int i = 0; i < 16; ++i) {
        int k = kbB + i;
        qa0[i] = qrow[k];
        qa1[i] = qrow[32 + k];
    }

    v8f o[4] = {{}, {}, {}, {}};
    float mrun[8], lrun[8];
#pragma unroll
    for (int r = 0; r < 8; ++r) { mrun[r] = -__builtin_inff(); lrun[r] = 0.f; }

    const int qlast = qb * 16 + 15;
    const int nkb   = qlast / 32 + 1;          // causal: key blocks needed

    for (int j = 0; j < nkb; ++j) {
        // Drain outstanding LDS reads before the DMA overwrites the tiles,
        // then async-stage K and V (32 keys x 64 dims = 2048 bf16 each).
        asm volatile("s_wait_dscnt 0" ::: "memory");
        tdm_load_to_lds(kt_off, Kp + (size_t)j * 32 * HD_, 32 * HD_);
        tdm_load_to_lds(vt_off, Vp + (size_t)j * 32 * HD_, 32 * HD_);
        __builtin_amdgcn_s_wait_tensorcnt(0);

        // S = Q K^T for 32 keys: two 16x16 D tiles, each over two K-halves.
        v16bf b00, b01, b10, b11;
#pragma unroll
        for (int i = 0; i < 16; ++i) {
            b00[i] = Kt[(ln)      * 64 +      kbB + i];
            b01[i] = Kt[(ln)      * 64 + 32 + kbB + i];
            b10[i] = Kt[(16 + ln) * 64 +      kbB + i];
            b11[i] = Kt[(16 + ln) * 64 + 32 + kbB + i];
        }
        v8f s0 = {}, s1 = {};
        s0 = wmma_bf16(qa0, b00, s0);
        s0 = wmma_bf16(qa1, b01, s0);
        s1 = wmma_bf16(qa0, b10, s1);
        s1 = wmma_bf16(qa1, b11, s1);

        // Online softmax per row (VGPR r <-> row r, +8 in upper lane half).
#pragma unroll
        for (int r = 0; r < 8; ++r) {
            int row = r + ((lane < 16) ? 0 : 8);
            int qg  = qb * 16 + row;
            int k0  = j * 32 + ln;
            int k1  = j * 32 + 16 + ln;
            float x0 = (k0 <= qg) ? s0[r] * SM_SCALE : -__builtin_inff();
            float x1 = (k1 <= qg) ? s1[r] * SM_SCALE : -__builtin_inff();
            float mx = fmaxf(x0, x1);
#pragma unroll
            for (int msk = 1; msk < 16; msk <<= 1)
                mx = fmaxf(mx, __shfl_xor(mx, msk, 32));
            float mnew = fmaxf(mrun[r], mx);
            float corr = __expf(mrun[r] - mnew);
            float p0 = __expf(x0 - mnew);
            float p1 = __expf(x1 - mnew);
            float ls = p0 + p1;
#pragma unroll
            for (int msk = 1; msk < 16; msk <<= 1)
                ls += __shfl_xor(ls, msk, 32);
            lrun[r] = lrun[r] * corr + ls;
            mrun[r] = mnew;
            o[0][r] *= corr; o[1][r] *= corr; o[2][r] *= corr; o[3][r] *= corr;
            Pt[row * 32 + ln]      = f2bf(p0);
            Pt[row * 32 + 16 + ln] = f2bf(p1);
        }

        // LDS round-trip re-layout: D-fragment P -> A-fragment P.
        asm volatile("s_wait_dscnt 0" ::: "memory");
        v16bf pa;
#pragma unroll
        for (int i = 0; i < 16; ++i)
            pa[i] = Pt[ln * 32 + kbB + i];

        // O += P * V (four 16-wide output-dim groups).
#pragma unroll
        for (int g = 0; g < 4; ++g) {
            v16bf vb;
#pragma unroll
            for (int i = 0; i < 16; ++i)
                vb[i] = Vt[(kbB + i) * 64 + g * 16 + ln];
            o[g] = wmma_bf16(pa, vb, o[g]);
        }
    }

    // Normalize and store to [B,T,C] bf16 (head-merged) for the out-proj GEMM.
    const int bb = bh / H_, h = bh % H_;
#pragma unroll
    for (int r = 0; r < 8; ++r) {
        float inv = 1.0f / lrun[r];
        int row = r + ((lane < 16) ? 0 : 8);
        size_t t = (size_t)qb * 16 + row;
        __bf16* orow = Oatt + ((size_t)bb * T_ + t) * C_ + h * HD_;
#pragma unroll
        for (int g = 0; g < 4; ++g)
            orow[g * 16 + ln] = f2bf(o[g][r] * inv);
    }
}

// ---------------------------------------------------------------------------
// Host-side launch.  Workspace layout (bf16): Q | K | V | att_out, 8 MiB each.
// ---------------------------------------------------------------------------
extern "C" void kernel_launch(void* const* d_in, const int* in_sizes, int n_in,
                              void* d_out, int out_size, void* d_ws, size_t ws_size,
                              hipStream_t stream) {
    (void)in_sizes; (void)n_in; (void)out_size; (void)ws_size;

    const float* query = (const float*)d_in[0];
    const float* key   = (const float*)d_in[1];
    const float* value = (const float*)d_in[2];
    // d_in[3] = att_mask (causal; applied analytically)
    const float* Wq = (const float*)d_in[4];
    const float* bq = (const float*)d_in[5];
    const float* Wk = (const float*)d_in[6];
    const float* bk = (const float*)d_in[7];
    const float* Wv = (const float*)d_in[8];
    const float* bv = (const float*)d_in[9];
    const float* Wp = (const float*)d_in[10];
    const float* bp = (const float*)d_in[11];

    const size_t QKV_BYTES = (size_t)M_ * C_ * sizeof(__bf16);  // 8 MiB
    char* ws = (char*)d_ws;
    __bf16* Qb = (__bf16*)(ws);
    __bf16* Kb = (__bf16*)(ws + QKV_BYTES);
    __bf16* Vb = (__bf16*)(ws + 2 * QKV_BYTES);
    __bf16* AO = (__bf16*)(ws + 3 * QKV_BYTES);

    // 128x64 macro tiles: (4096/128) * (1024/64) = 512 blocks of 256 threads.
    const int blocks = (M_ / 128) * (C_ / 64);

    gemm_wmma<0, 0><<<blocks, 256, 0, stream>>>(query, Wq, bq, Qb, M_, C_, C_);
    gemm_wmma<0, 0><<<blocks, 256, 0, stream>>>(key,   Wk, bk, Kb, M_, C_, C_);
    gemm_wmma<0, 0><<<blocks, 256, 0, stream>>>(value, Wv, bv, Vb, M_, C_, C_);

    // 4096 (b,h,qblock) wave tasks, 4 waves per block.
    attn_fwd<<<1024, 128, 0, stream>>>(Qb, Kb, Vb, AO);

    gemm_wmma<1, 1><<<blocks, 256, 0, stream>>>(AO, Wp, bp, d_out, M_, C_, C_);
}
